// TFBlipAttention_50706383896955
// MI455X (gfx1250) — matrix operand order repaired and checked by reading, not compiled, not run on hardware
//
#include <hip/hip_runtime.h>

// ---------------------------------------------------------------------------
// BLIP attention block on gfx1250 (MI455X), bf16 WMMA everywhere.
//   B=8 T=1024 E=768 H=12 Dh=64  (~65 GFLOP, compute-bound -> WMMA bf16)
// Round 2: double-buffered 32x64 GEMM tiles, prefetched K frags in attention,
// SCALE*log2e folded into Q, exp2-based online softmax.
// ---------------------------------------------------------------------------

typedef __attribute__((ext_vector_type(16))) __bf16 v16bf;
typedef __attribute__((ext_vector_type(8)))  __bf16 bf16x8;
typedef __attribute__((ext_vector_type(8)))  float  v8f;

#define EMB   768
#define T3E   2304
#define NTOK  8192      // B*T
#define NBH   96        // B*H
// SCALE * log2(e) = 0.125 * 1.4426950408889634
#define QSCALE 0.18033688011112042f

// round-to-nearest-even f32 -> bf16 (stored as ushort)
static __device__ __forceinline__ unsigned short f2bf(float f) {
  unsigned u = __float_as_uint(f);
  return (unsigned short)((u + 0x7FFFu + ((u >> 16) & 1u)) >> 16);
}

// Build a 16-half WMMA fragment from two 16B vector loads.
static __device__ __forceinline__ v16bf mkfrag(const unsigned short* lo,
                                               const unsigned short* hi) {
  bf16x8 a = *reinterpret_cast<const bf16x8*>(lo);
  bf16x8 b = *reinterpret_cast<const bf16x8*>(hi);
  v16bf r;
#pragma unroll
  for (int i = 0; i < 8; ++i) { r[i] = a[i]; r[8 + i] = b[i]; }
  return r;
}

static __device__ __forceinline__ v8f wmma_bf16(v16bf a, v16bf b, v8f c) {
  return __builtin_amdgcn_wmma_f32_16x16x32_bf16(
      false, a, false, b, (short)0, c, false, false);
}

// --------------------------- prep kernels ----------------------------------

__global__ void cvt_f32_bf16(const float* __restrict__ in,
                             unsigned short* __restrict__ out, int n) {
  for (int i = blockIdx.x * blockDim.x + threadIdx.x; i < n;
       i += gridDim.x * blockDim.x)
    out[i] = f2bf(in[i]);
}

// in[rows][cols] f32 -> out[cols][rows] bf16  (W -> W^T for B-fragment loads)
__global__ void transpose_f32_bf16(const float* __restrict__ in,
                                   unsigned short* __restrict__ out,
                                   int rows, int cols) {
  int n = rows * cols;
  for (int i = blockIdx.x * blockDim.x + threadIdx.x; i < n;
       i += gridDim.x * blockDim.x) {
    int r = i / cols, c = i % cols;
    out[(size_t)c * rows + r] = f2bf(in[i]);
  }
}

// --------------------------- QKV GEMM --------------------------------------
// C[8192,2304] = X[8192,768] @ Wqkv ; one wave -> 32x64 tile, double-buffered.
// Epilogue scatters Q*QSCALE, K [B,H,T,64] and V^T [B,H,64,T] (bf16).
__global__ __launch_bounds__(256) void qkv_gemm(
    const unsigned short* __restrict__ X, const unsigned short* __restrict__ WT,
    const float* __restrict__ bias, unsigned short* __restrict__ Q,
    unsigned short* __restrict__ K, unsigned short* __restrict__ VT) {
  const int lane = threadIdx.x & 31;
  const int wave = blockIdx.x * 8 + (threadIdx.x >> 5);
  const int mTile = wave / 36, nTile = wave % 36;   // 256 x 36 tiles
  const int m0 = mTile * 32, n0 = nTile * 64;
  const int grp = lane >> 4, lr = lane & 15;

  v8f z = {};
  v8f accLo[4], accHi[4];
#pragma unroll
  for (int s = 0; s < 4; ++s) { accLo[s] = z; accHi[s] = z; }

  const unsigned short* Arow0 = X + (size_t)(m0 + lr) * EMB + grp * 16;
  const unsigned short* Arow1 = Arow0 + (size_t)16 * EMB;
  const unsigned short* Brow[4];
#pragma unroll
  for (int s = 0; s < 4; ++s)
    Brow[s] = WT + (size_t)(n0 + s * 16 + lr) * EMB + grp * 16;

  // preload kk = 0
  v16bf a0 = mkfrag(Arow0, Arow0 + 8);
  v16bf a1 = mkfrag(Arow1, Arow1 + 8);
  v16bf b0 = mkfrag(Brow[0], Brow[0] + 8);
  v16bf b1 = mkfrag(Brow[1], Brow[1] + 8);
  v16bf b2 = mkfrag(Brow[2], Brow[2] + 8);
  v16bf b3 = mkfrag(Brow[3], Brow[3] + 8);

#pragma unroll
  for (int kk = 0; kk < EMB; kk += 32) {
    const int kn = kk + 32;
    v16bf a0n, a1n, b0n, b1n, b2n, b3n;
    if (kn < EMB) {   // issue next-step loads before the current WMMAs
      a0n = mkfrag(Arow0 + kn, Arow0 + kn + 8);
      a1n = mkfrag(Arow1 + kn, Arow1 + kn + 8);
      b0n = mkfrag(Brow[0] + kn, Brow[0] + kn + 8);
      b1n = mkfrag(Brow[1] + kn, Brow[1] + kn + 8);
      b2n = mkfrag(Brow[2] + kn, Brow[2] + kn + 8);
      b3n = mkfrag(Brow[3] + kn, Brow[3] + kn + 8);
    }
    accLo[0] = wmma_bf16(a0, b0, accLo[0]);
    accHi[0] = wmma_bf16(a1, b0, accHi[0]);
    accLo[1] = wmma_bf16(a0, b1, accLo[1]);
    accHi[1] = wmma_bf16(a1, b1, accHi[1]);
    accLo[2] = wmma_bf16(a0, b2, accLo[2]);
    accHi[2] = wmma_bf16(a1, b2, accHi[2]);
    accLo[3] = wmma_bf16(a0, b3, accLo[3]);
    accHi[3] = wmma_bf16(a1, b3, accHi[3]);
    if (kn < EMB) {
      a0 = a0n; a1 = a1n; b0 = b0n; b1 = b1n; b2 = b2n; b3 = b3n;
    }
  }

#pragma unroll
  for (int s = 0; s < 4; ++s) {
    const int c = n0 + s * 16 + lr;        // 16-block never crosses 768/64 edge
    const int which = c / EMB, rem = c % EMB;
    const int h = rem / 64, d = rem % 64;
    const float bv = bias[c];
#pragma unroll
    for (int half = 0; half < 2; ++half) {
#pragma unroll
      for (int r = 0; r < 8; ++r) {
        const int tg = m0 + half * 16 + grp * 8 + r;   // global token
        const int b_ = tg >> 10, t = tg & 1023;
        const float val = (half ? accHi[s][r] : accLo[s][r]) + bv;
        const size_t bh = (size_t)(b_ * 12 + h);
        if (which == 0)      Q[(bh * 1024 + t) * 64 + d] = f2bf(val * QSCALE);
        else if (which == 1) K[(bh * 1024 + t) * 64 + d] = f2bf(val);
        else                 VT[(bh * 64 + d) * 1024 + t] = f2bf(val);
      }
    }
  }
}

// --------------------------- flash attention -------------------------------
// One wave per (b,h, 16-query tile). 32 keys/step, exp2-based online softmax.
// Q is pre-scaled by SCALE*log2(e); softmax over 2^s' == softmax over exp(s).
__global__ __launch_bounds__(128) void attn_kernel(
    const unsigned short* __restrict__ Q, const unsigned short* __restrict__ K,
    const unsigned short* __restrict__ VT, unsigned short* __restrict__ Ctx) {
  __shared__ unsigned short pbuf[4][16 * 32];   // per-wave P tile (bf16)
  const int lane = threadIdx.x & 31;
  const int wid = threadIdx.x >> 5;
  const int wave = blockIdx.x * 4 + wid;        // 6144 waves total
  const int bh = wave >> 6;                     // 0..95
  const int qt = wave & 63;                     // query tile 0..63
  const int grp = lane >> 4, lr = lane & 15;

  const unsigned short* Qp = Q + (size_t)bh * (1024 * 64);
  const unsigned short* Kp = K + (size_t)bh * (1024 * 64);
  const unsigned short* Vp = VT + (size_t)bh * (64 * 1024);

  const unsigned short* qrow = Qp + (size_t)(qt * 16 + lr) * 64;
  const v16bf qa0 = mkfrag(qrow + grp * 16, qrow + 8 + grp * 16);       // Dh 0..31
  const v16bf qa1 = mkfrag(qrow + 32 + grp * 16, qrow + 40 + grp * 16);  // Dh 32..63

  v8f o0 = {}, o1 = {}, o2 = {}, o3 = {};
  float m[8], l[8];
#pragma unroll
  for (int r = 0; r < 8; ++r) { m[r] = -3.0e38f; l[r] = 0.f; }
  unsigned short* pl = pbuf[wid];

  // preload K fragments for j = 0
  const unsigned short* k0 = Kp + (size_t)lr * 64 + grp * 16;
  const unsigned short* k1 = Kp + (size_t)(16 + lr) * 64 + grp * 16;
  v16bf kb00 = mkfrag(k0, k0 + 8), kb01 = mkfrag(k0 + 32, k0 + 40);
  v16bf kb10 = mkfrag(k1, k1 + 8), kb11 = mkfrag(k1 + 32, k1 + 40);

  for (int j = 0; j < 1024; j += 32) {
    // V fragments for this step (in flight through scores + softmax)
    const unsigned short* vbase = Vp + (size_t)lr * 1024 + j + grp * 16;
    const v16bf vf0 = mkfrag(vbase, vbase + 8);
    const v16bf vf1 = mkfrag(vbase + 16 * 1024, vbase + 16 * 1024 + 8);
    const v16bf vf2 = mkfrag(vbase + 32 * 1024, vbase + 32 * 1024 + 8);
    const v16bf vf3 = mkfrag(vbase + 48 * 1024, vbase + 48 * 1024 + 8);

    // ---- scores: Q(16x64) @ K^T for 32 keys -> two 16x16 C fragments
    v8f s0 = {}, s1 = {};
    s0 = wmma_bf16(qa0, kb00, s0);
    s0 = wmma_bf16(qa1, kb01, s0);
    s1 = wmma_bf16(qa0, kb10, s1);
    s1 = wmma_bf16(qa1, kb11, s1);

    // ---- prefetch K fragments for next step (hidden behind softmax VALU)
    if (j + 32 < 1024) {
      const unsigned short* n0 = Kp + (size_t)(j + 32 + lr) * 64 + grp * 16;
      const unsigned short* n1 = Kp + (size_t)(j + 48 + lr) * 64 + grp * 16;
      kb00 = mkfrag(n0, n0 + 8); kb01 = mkfrag(n0 + 32, n0 + 40);
      kb10 = mkfrag(n1, n1 + 8); kb11 = mkfrag(n1 + 32, n1 + 40);
    }

    // ---- online softmax (row reductions across 16-lane half groups)
#pragma unroll
    for (int r = 0; r < 8; ++r) {
      const float sa = s0[r], sb = s1[r];   // already scaled (folded into Q)
      float mx = fmaxf(sa, sb);
      mx = fmaxf(mx, __shfl_xor(mx, 1, 32));
      mx = fmaxf(mx, __shfl_xor(mx, 2, 32));
      mx = fmaxf(mx, __shfl_xor(mx, 4, 32));
      mx = fmaxf(mx, __shfl_xor(mx, 8, 32));
      const float mn = fmaxf(m[r], mx);
      const float alpha = __builtin_amdgcn_exp2f(m[r] - mn);
      const float pa = __builtin_amdgcn_exp2f(sa - mn);
      const float pb = __builtin_amdgcn_exp2f(sb - mn);
      float rs = pa + pb;
      rs += __shfl_xor(rs, 1, 32);
      rs += __shfl_xor(rs, 2, 32);
      rs += __shfl_xor(rs, 4, 32);
      rs += __shfl_xor(rs, 8, 32);
      m[r] = mn;
      l[r] = l[r] * alpha + rs;
      o0[r] *= alpha; o1[r] *= alpha; o2[r] *= alpha; o3[r] *= alpha;
      // P tile row-major [16][32] in LDS (C layout -> A layout shuffle)
      pl[(grp * 8 + r) * 32 + lr] = f2bf(pa);
      pl[(grp * 8 + r) * 32 + 16 + lr] = f2bf(pb);
    }
    asm volatile("s_wait_dscnt 0x0" ::: "memory");  // DS store->load, same wave

    const v16bf pf = mkfrag(pl + lr * 32 + grp * 16, pl + lr * 32 + 8 + grp * 16);

    // ---- O += P(16x32) @ V(32x64); V^T rows are contiguous in key index
    o0 = wmma_bf16(pf, vf0, o0);
    o1 = wmma_bf16(pf, vf1, o1);
    o2 = wmma_bf16(pf, vf2, o2);
    o3 = wmma_bf16(pf, vf3, o3);
  }

  // ---- epilogue: ctx[b][t][h*64+d] bf16
  const int b_ = bh / 12, h = bh % 12;
#pragma unroll
  for (int r = 0; r < 8; ++r) {
    const float inv = 1.0f / l[r];
    const size_t row =
        (size_t)(b_ * 1024 + qt * 16 + grp * 8 + r) * EMB + h * 64;
    Ctx[row + 0 + lr]  = f2bf(o0[r] * inv);
    Ctx[row + 16 + lr] = f2bf(o1[r] * inv);
    Ctx[row + 32 + lr] = f2bf(o2[r] * inv);
    Ctx[row + 48 + lr] = f2bf(o3[r] * inv);
  }
}

// --------------------------- output projection -----------------------------
// out[8192,768] f32 = Ctx[8192,768]bf16 @ proj_w + proj_b ; 32x64 tiles.
__global__ __launch_bounds__(256) void proj_gemm(
    const unsigned short* __restrict__ Ctx, const unsigned short* __restrict__ WT,
    const float* __restrict__ bias, float* __restrict__ out) {
  const int lane = threadIdx.x & 31;
  const int wave = blockIdx.x * 8 + (threadIdx.x >> 5);
  const int mTile = wave / 12, nTile = wave % 12;   // 256 x 12 tiles
  const int m0 = mTile * 32, n0 = nTile * 64;
  const int grp = lane >> 4, lr = lane & 15;

  v8f z = {};
  v8f accLo[4], accHi[4];
#pragma unroll
  for (int s = 0; s < 4; ++s) { accLo[s] = z; accHi[s] = z; }

  const unsigned short* Arow0 = Ctx + (size_t)(m0 + lr) * EMB + grp * 16;
  const unsigned short* Arow1 = Arow0 + (size_t)16 * EMB;
  const unsigned short* Brow[4];
#pragma unroll
  for (int s = 0; s < 4; ++s)
    Brow[s] = WT + (size_t)(n0 + s * 16 + lr) * EMB + grp * 16;

  v16bf a0 = mkfrag(Arow0, Arow0 + 8);
  v16bf a1 = mkfrag(Arow1, Arow1 + 8);
  v16bf b0 = mkfrag(Brow[0], Brow[0] + 8);
  v16bf b1 = mkfrag(Brow[1], Brow[1] + 8);
  v16bf b2 = mkfrag(Brow[2], Brow[2] + 8);
  v16bf b3 = mkfrag(Brow[3], Brow[3] + 8);

#pragma unroll
  for (int kk = 0; kk < EMB; kk += 32) {
    const int kn = kk + 32;
    v16bf a0n, a1n, b0n, b1n, b2n, b3n;
    if (kn < EMB) {
      a0n = mkfrag(Arow0 + kn, Arow0 + kn + 8);
      a1n = mkfrag(Arow1 + kn, Arow1 + kn + 8);
      b0n = mkfrag(Brow[0] + kn, Brow[0] + kn + 8);
      b1n = mkfrag(Brow[1] + kn, Brow[1] + kn + 8);
      b2n = mkfrag(Brow[2] + kn, Brow[2] + kn + 8);
      b3n = mkfrag(Brow[3] + kn, Brow[3] + kn + 8);
    }
    accLo[0] = wmma_bf16(a0, b0, accLo[0]);
    accHi[0] = wmma_bf16(a1, b0, accHi[0]);
    accLo[1] = wmma_bf16(a0, b1, accLo[1]);
    accHi[1] = wmma_bf16(a1, b1, accHi[1]);
    accLo[2] = wmma_bf16(a0, b2, accLo[2]);
    accHi[2] = wmma_bf16(a1, b2, accHi[2]);
    accLo[3] = wmma_bf16(a0, b3, accLo[3]);
    accHi[3] = wmma_bf16(a1, b3, accHi[3]);
    if (kn < EMB) {
      a0 = a0n; a1 = a1n; b0 = b0n; b1 = b1n; b2 = b2n; b3 = b3n;
    }
  }

#pragma unroll
  for (int s = 0; s < 4; ++s) {
    const int c = n0 + s * 16 + lr;
    const float bv = bias[c];
#pragma unroll
    for (int r = 0; r < 8; ++r) {
      out[(size_t)(m0 + grp * 8 + r) * EMB + c] = accLo[s][r] + bv;
      out[(size_t)(m0 + 16 + grp * 8 + r) * EMB + c] = accHi[s][r] + bv;
    }
  }
}

// --------------------------- launcher --------------------------------------
extern "C" void kernel_launch(void* const* d_in, const int* in_sizes, int n_in,
                              void* d_out, int out_size, void* d_ws,
                              size_t ws_size, hipStream_t stream) {
  const float* hs     = (const float*)d_in[0];  // [8,1024,768]
  const float* qkv_w  = (const float*)d_in[1];  // [768,2304]
  const float* qkv_b  = (const float*)d_in[2];  // [2304]
  const float* proj_w = (const float*)d_in[3];  // [768,768]
  const float* proj_b = (const float*)d_in[4];  // [768]
  float* out = (float*)d_out;

  // workspace layout (bf16 stored as ushort), total ~67.6 MB
  unsigned short* Xbf    = (unsigned short*)d_ws;
  unsigned short* WqkvT  = Xbf   + (size_t)NTOK * EMB;        // [2304][768]
  unsigned short* WprojT = WqkvT + (size_t)T3E * EMB;         // [768][768]
  unsigned short* Qb     = WprojT + (size_t)EMB * EMB;        // [96][1024][64]
  unsigned short* Kb     = Qb    + (size_t)NBH * 1024 * 64;
  unsigned short* VTb    = Kb    + (size_t)NBH * 1024 * 64;   // [96][64][1024]
  unsigned short* Ctx    = VTb   + (size_t)NBH * 1024 * 64;   // [8192][768]

  cvt_f32_bf16<<<1024, 256, 0, stream>>>(hs, Xbf, NTOK * EMB);
  transpose_f32_bf16<<<1024, 256, 0, stream>>>(qkv_w, WqkvT, EMB, T3E);
  transpose_f32_bf16<<<512, 256, 0, stream>>>(proj_w, WprojT, EMB, EMB);

  // 256*36 = 9216 waves, 8 waves/block
  qkv_gemm<<<1152, 256, 0, stream>>>(Xbf, WqkvT, qkv_b, Qb, Kb, VTb);
  // 96*64 = 6144 waves, 4 waves/block
  attn_kernel<<<1536, 128, 0, stream>>>(Qb, Kb, VTb, Ctx);
  // 256*12 = 3072 waves, 8 waves/block
  proj_gemm<<<384, 256, 0, stream>>>(Ctx, WprojT, proj_b, out);
}
